// TernBinLayer_20787641712654
// MI455X (gfx1250) — 8-bit matrix operand order repaired and checked by reading, compiled, not run on hardware
//
#include <hip/hip_runtime.h>

typedef __attribute__((ext_vector_type(2))) int v2i;
typedef __attribute__((ext_vector_type(4))) int v4i;
typedef __attribute__((ext_vector_type(8))) int v8i;
typedef __attribute__((ext_vector_type(4))) float v4f;

#define M_DIM 8192
#define K_DIM 4096
#define N_DIM 4096

#define BM 128
#define BN 128
#define BK 64
#define LDSS 80

#if defined(__gfx1250__) && __has_builtin(__builtin_amdgcn_global_load_async_to_lds_b128)
#define HAS_ASYNC_LDS 1
#else
#define HAS_ASYNC_LDS 0
#endif

__global__ __launch_bounds__(256) void cvt_x_kernel(const float* __restrict__ x,
                                                    char* __restrict__ xi) {
  const size_t base = ((size_t)blockIdx.x * 256 + threadIdx.x) * 16;
  int packed[4];
#pragma unroll
  for (int i = 0; i < 4; ++i) {
    v4f f = __builtin_nontemporal_load((const v4f*)(x + base + i * 4));
    int b0 = (f.x > 0.0f) ? 0x01 : 0xFF;
    int b1 = (f.y > 0.0f) ? 0x01 : 0xFF;
    int b2 = (f.z > 0.0f) ? 0x01 : 0xFF;
    int b3 = (f.w > 0.0f) ? 0x01 : 0xFF;
    packed[i] = b0 | (b1 << 8) | (b2 << 16) | (b3 << 24);
  }
  v4i v = {packed[0], packed[1], packed[2], packed[3]};
  *(v4i*)(xi + base) = v;
}

__global__ __launch_bounds__(256) void cvt_w_kernel(const float* __restrict__ W,
                                                    char* __restrict__ Wt) {
  __shared__ char tile[32][36];
  const int n0 = blockIdx.x * 32;
  const int k0 = blockIdx.y * 32;
  const int tx = threadIdx.x & 31;
  const int ty = threadIdx.x >> 5;
#pragma unroll
  for (int i = 0; i < 4; ++i) {
    const int k = ty + i * 8;
    const float f = __builtin_nontemporal_load(&W[(size_t)(k0 + k) * N_DIM + (n0 + tx)]);
    tile[tx][k] = (f > 0.5f) ? (char)1 : ((f < -0.5f) ? (char)-1 : (char)0);
  }
  __syncthreads();
  const int nl = threadIdx.x >> 3;
  const int kq = threadIdx.x & 7;
  const int v = *(const int*)&tile[nl][kq * 4];
  *(int*)(Wt + (size_t)(n0 + nl) * K_DIM + k0 + kq * 4) = v;
}

#if HAS_ASYNC_LDS
typedef __attribute__((address_space(1))) v4i gv4i;
typedef __attribute__((address_space(3))) v4i sv4i;

__device__ __forceinline__ void async_cp16(const char* g, char* l) {
  __builtin_amdgcn_global_load_async_to_lds_b128(
      (gv4i*)(char*)g, (sv4i*)l, 0, 0);
}
#endif

__global__ __launch_bounds__(256) void bnn_gemm_kernel(const char* __restrict__ A,
                                                       const char* __restrict__ B,
                                                       float* __restrict__ out) {
  constexpr int ABUF = BM * LDSS;
  __shared__ char sA[2 * ABUF];
  __shared__ char sB[2 * ABUF];

  const int tid = threadIdx.x;
  const int wave = tid >> 5;
  const int lane = tid & 31;
  const int m0 = blockIdx.y * BM;
  const int n0 = blockIdx.x * BN;

  const int ldRow = tid >> 1;
  const int ldCol = (tid & 1) * 32;
  const char* gA = A + (size_t)(m0 + ldRow) * K_DIM + ldCol;
  const char* gB = B + (size_t)(n0 + ldRow) * K_DIM + ldCol;
  const int sOff = ldRow * LDSS + ldCol;

  const int wm = (wave & 3) * 32;
  const int wn = (wave >> 2) * 64;
  const int hi = lane >> 4;
  const int l15 = lane & 15;

  const int aFragOff = (wm + l15) * LDSS + (hi ? 16 : 0);
  const int bFragOff = (wn + l15) * LDSS + (hi ? 16 : 0);

  v8i zero = {0, 0, 0, 0, 0, 0, 0, 0};
  v8i acc[2][4];
#pragma unroll
  for (int i = 0; i < 2; ++i)
#pragma unroll
    for (int j = 0; j < 4; ++j) acc[i][j] = zero;

#if HAS_ASYNC_LDS
  async_cp16(gA, &sA[sOff]);
  async_cp16(gA + 16, &sA[sOff + 16]);
  async_cp16(gB, &sB[sOff]);
  async_cp16(gB + 16, &sB[sOff + 16]);
  asm volatile("s_wait_asynccnt 0x0" ::: "memory");
#else
  *(v4i*)(&sA[sOff])      = *(const v4i*)(gA);
  *(v4i*)(&sA[sOff + 16]) = *(const v4i*)(gA + 16);
  *(v4i*)(&sB[sOff])      = *(const v4i*)(gB);
  *(v4i*)(&sB[sOff + 16]) = *(const v4i*)(gB + 16);
#endif
  __syncthreads();

  unsigned rdO = 0, wrO = ABUF;
  const int KT = K_DIM / BK;

#pragma clang loop unroll(disable)
  for (int kt = 0; kt < KT; ++kt) {
#if !HAS_ASYNC_LDS
    v4i ra0, ra1, rb0, rb1;
#endif
    if (kt + 1 < KT) {
      const size_t off = (size_t)(kt + 1) * BK;
#if HAS_ASYNC_LDS
      async_cp16(gA + off,      &sA[wrO + sOff]);
      async_cp16(gA + off + 16, &sA[wrO + sOff + 16]);
      async_cp16(gB + off,      &sB[wrO + sOff]);
      async_cp16(gB + off + 16, &sB[wrO + sOff + 16]);
#else
      ra0 = *(const v4i*)(gA + off);
      ra1 = *(const v4i*)(gA + off + 16);
      rb0 = *(const v4i*)(gB + off);
      rb1 = *(const v4i*)(gB + off + 16);
#endif
    }

    v8i af[2];
#pragma unroll
    for (int mi = 0; mi < 2; ++mi) {
      const char* p = &sA[rdO + aFragOff + mi * 16 * LDSS];
      v4i t0 = *(const v4i*)(p);
      v4i t1 = *(const v4i*)(p + 32);
      v8i a = {t0.x, t0.y, t0.z, t0.w, t1.x, t1.y, t1.z, t1.w};
      af[mi] = a;
    }
    v8i bf[4];
#pragma unroll
    for (int ni = 0; ni < 4; ++ni) {
      const char* p = &sB[rdO + bFragOff + ni * 16 * LDSS];
      v4i t0 = *(const v4i*)(p);
      v4i t1 = *(const v4i*)(p + 32);
      v8i b = {t0.x, t0.y, t0.z, t0.w, t1.x, t1.y, t1.z, t1.w};
      bf[ni] = b;
    }

#pragma unroll
    for (int mi = 0; mi < 2; ++mi)
#pragma unroll
      for (int ni = 0; ni < 4; ++ni)
        acc[mi][ni] = __builtin_amdgcn_wmma_i32_16x16x64_iu8(
             true, af[mi],  true, bf[ni], acc[mi][ni],
             false,  false);

#if HAS_ASYNC_LDS
    if (kt + 1 < KT) {
      asm volatile("s_wait_asynccnt 0x0" ::: "memory");
      __syncthreads();
    }
#else
    __syncthreads();
    if (kt + 1 < KT) {
      *(v4i*)(&sA[wrO + sOff])      = ra0;
      *(v4i*)(&sA[wrO + sOff + 16]) = ra1;
      *(v4i*)(&sB[wrO + sOff])      = rb0;
      *(v4i*)(&sB[wrO + sOff + 16]) = rb1;
      __syncthreads();
    }
#endif
    const unsigned t = rdO; rdO = wrO; wrO = t;
  }

#pragma unroll
  for (int mi = 0; mi < 2; ++mi) {
#pragma unroll
    for (int ni = 0; ni < 4; ++ni) {
#pragma unroll
      for (int r = 0; r < 8; ++r) {
        const int row = m0 + wm + mi * 16 + r + hi * 8;
        const int col = n0 + wn + ni * 16 + l15;
        const float v = (acc[mi][ni][r] > 0) ? 1.0f : -1.0f;
        __builtin_nontemporal_store(v, &out[(size_t)row * N_DIM + col]);
      }
    }
  }
}

extern "C" void kernel_launch(void* const* d_in, const int* in_sizes, int n_in,
                              void* d_out, int out_size, void* d_ws, size_t ws_size,
                              hipStream_t stream) {
  (void)in_sizes; (void)n_in; (void)out_size; (void)ws_size;
  const float* x = (const float*)d_in[0];
  const float* W = (const float*)d_in[1];
  float* out = (float*)d_out;

  char* xi = (char*)d_ws;
  char* Wt = (char*)d_ws + (size_t)M_DIM * K_DIM;

  cvt_x_kernel<<<(M_DIM * K_DIM) / (256 * 16), 256, 0, stream>>>(x, xi);
  cvt_w_kernel<<<dim3(N_DIM / 32, K_DIM / 32), 256, 0, stream>>>(W, Wt);
  bnn_gemm_kernel<<<dim3(N_DIM / BN, M_DIM / BM), 256, 0, stream>>>(xi, Wt, out);
}
